// SelectiveSSM_82910048682684
// MI455X (gfx1250) — matrix operand order repaired and checked by reading, not compiled, not run on hardware
//
#include <hip/hip_runtime.h>
#include <hip/hip_bf16.h>

// ---------------------------------------------------------------------------
// Problem constants (from reference): B=4, L=2048, D_MODEL=1024, D_INNER=2048,
// D_STATE=16, D_CONV=4.
// ---------------------------------------------------------------------------
#define BATCH   4
#define SEQ     2048
#define DM      1024
#define DI      2048
#define NSTATE  16
#define BL      (BATCH * SEQ)       // 8192 rows

typedef __attribute__((ext_vector_type(16))) __bf16 v16bf;
typedef __attribute__((ext_vector_type(8)))  float  v8f;

static __device__ __forceinline__ unsigned short f2bf(float f) {
  unsigned int u = __float_as_uint(f);
  unsigned int r = u + 0x7FFFu + ((u >> 16) & 1u);   // round-to-nearest-even
  return (unsigned short)(r >> 16);
}

// ---------------------------------------------------------------------------
// Async global->LDS copy (GLOBAL_LOAD_ASYNC_TO_LDS_B128, tracked by ASYNCcnt).
// Builtin signature (probed from clang diagnostics): param0 is a non-const
// AS1 pointer to a 16-byte int vector; param1 the AS3 counterpart.
// LDS address is the low 32 bits of the generic pointer (AS3 ptrs are 32-bit).
// ---------------------------------------------------------------------------
typedef int v4i_gcc __attribute__((vector_size(16)));
typedef __attribute__((address_space(1))) v4i_gcc* as1_v4i_ptr;
typedef __attribute__((address_space(3))) v4i_gcc* as3_v4i_ptr;

#if __has_builtin(__builtin_amdgcn_global_load_async_to_lds_b128)
#define ASYNC_COPY_B128(gp, laddr)                                             \
  __builtin_amdgcn_global_load_async_to_lds_b128(                              \
      (as1_v4i_ptr)(unsigned long long)(uintptr_t)(gp),                        \
      (as3_v4i_ptr)(unsigned)(laddr), 0, 0)
#else
#define ASYNC_COPY_B128(gp, laddr)                                             \
  asm volatile("global_load_async_to_lds_b128 %0, %1, off" ::                  \
                   "v"((unsigned)(laddr)),                                     \
                   "v"((const void*)(gp))                                      \
               : "memory")
#endif

#if __has_builtin(__builtin_amdgcn_s_wait_asynccnt)
#define WAIT_ASYNC(n) __builtin_amdgcn_s_wait_asynccnt(n)
#else
#define WAIT_ASYNC(n) asm volatile("s_wait_asynccnt %0" ::"i"(n))
#endif

// ---------------------------------------------------------------------------
// Elementwise f32 -> bf16 convert
// ---------------------------------------------------------------------------
__global__ void cvt_bf16_kernel(const float* __restrict__ in,
                                unsigned short* __restrict__ out, long n) {
  long i = (long)blockIdx.x * blockDim.x + threadIdx.x;
  long stride = (long)gridDim.x * blockDim.x;
  for (; i < n; i += stride) out[i] = f2bf(in[i]);
}

// Transpose R x C (f32, row-major) -> C x R (bf16, row-major)
__global__ void transpose_bf16_kernel(const float* __restrict__ in,
                                      unsigned short* __restrict__ out,
                                      int R, int C) {
  long n = (long)R * C;
  long i = (long)blockIdx.x * blockDim.x + threadIdx.x;
  long stride = (long)gridDim.x * blockDim.x;
  for (; i < n; i += stride) {
    int c = (int)(i % C);
    int r = (int)(i / C);
    out[(long)c * R + r] = f2bf(in[i]);
  }
}

// ---------------------------------------------------------------------------
// BF16 WMMA GEMM: C(MxN, f32) = A(MxK, bf16 row-major) x Bt(NxK, bf16 row-major)
// Block tile 128x128, K-step 64, 256 threads = 8 waves in a 2(M) x 4(N) grid;
// each wave computes 4x2 tiles of 16x16 via v_wmma_f32_16x16x32_bf16.
// Double-buffered LDS filled with async global->LDS B128 copies (ASYNCcnt),
// prefetched one full K-step ahead.  M, N multiples of 128; K multiple of 64.
// ---------------------------------------------------------------------------
#define BMT 128
#define BNT 128
#define BKT 64
#define LSTR 72   // bf16 elements per LDS row (144 bytes = 9*16B, bank-spread)

__global__ __launch_bounds__(256) void gemm_bf16_wmma_kernel(
    const unsigned short* __restrict__ A,
    const unsigned short* __restrict__ Bt,
    float* __restrict__ C,
    int M, int N, int K) {
  __shared__ __align__(16) unsigned short As[2][BMT * LSTR];
  __shared__ __align__(16) unsigned short Bs[2][BNT * LSTR];

  const int tid  = threadIdx.x;
  const int lane = tid & 31;
  const int half = lane >> 4;     // 0: lanes 0-15, 1: lanes 16-31
  const int lr   = lane & 15;
  const int wave = tid >> 5;
  const int wm   = (wave & 1) * 64;   // wave row offset inside block tile
  const int wn   = (wave >> 1) * 32;  // wave col offset inside block tile
  const int bm   = blockIdx.y * BMT;
  const int bn   = blockIdx.x * BNT;

  v8f acc[4][2] = {};

  // Cooperative staging: 2 threads per row; each thread owns 4 contiguous
  // 16B chunks (= 32 bf16) of the 64-wide K slab, for both A and B.
  const int lrow = tid >> 1;          // 0..127
  const int ccol = (tid & 1) * 32;    // element column base within the slab

  const unsigned short* Agr = A  + (long)(bm + lrow) * K + ccol;
  const unsigned short* Bgr = Bt + (long)(bn + lrow) * K + ccol;

  // Per-tile issue: 8 async B128 copies per thread (4 for A, 4 for B).
#define ISSUE_TILE(k0, buf)                                                    \
  do {                                                                         \
    const unsigned short* ag = Agr + (k0);                                     \
    const unsigned short* bg = Bgr + (k0);                                     \
    unsigned la = (unsigned)(uintptr_t)&As[buf][lrow * LSTR + ccol];           \
    unsigned lb = (unsigned)(uintptr_t)&Bs[buf][lrow * LSTR + ccol];           \
    ASYNC_COPY_B128(ag,      la);                                              \
    ASYNC_COPY_B128(ag +  8, la + 16);                                         \
    ASYNC_COPY_B128(ag + 16, la + 32);                                         \
    ASYNC_COPY_B128(ag + 24, la + 48);                                         \
    ASYNC_COPY_B128(bg,      lb);                                              \
    ASYNC_COPY_B128(bg +  8, lb + 16);                                         \
    ASYNC_COPY_B128(bg + 16, lb + 32);                                         \
    ASYNC_COPY_B128(bg + 24, lb + 48);                                         \
  } while (0)

  const int nk = K / BKT;
  ISSUE_TILE(0, 0);

  for (int k = 0; k < nk; ++k) {
    const int cur = k & 1;
    if (k + 1 < nk) {
      ISSUE_TILE((k + 1) * BKT, cur ^ 1);
      WAIT_ASYNC(8);          // current tile landed; prefetch may be in flight
    } else {
      WAIT_ASYNC(0);
    }
    __syncthreads();

    const unsigned short* Ab = As[cur];
    const unsigned short* Bb = Bs[cur];
#pragma unroll
    for (int kk = 0; kk < BKT; kk += 32) {
      union Frag { uint4 u[2]; v16bf v; };
      Frag bf2[2], af[4];
#pragma unroll
      for (int ni = 0; ni < 2; ni++) {
        const uint4* p =
            (const uint4*)(Bb + (wn + ni * 16 + lr) * LSTR + kk + half * 16);
        bf2[ni].u[0] = p[0];
        bf2[ni].u[1] = p[1];
      }
#pragma unroll
      for (int mi = 0; mi < 4; mi++) {
        const unsigned short* ar =
            Ab + (wm + mi * 16 + lr) * LSTR + kk + half * 16;
        af[mi].u[0] = *(const uint4*)(ar);
        af[mi].u[1] = *(const uint4*)(ar + 8);
      }
#pragma unroll
      for (int mi = 0; mi < 4; mi++) {
#pragma unroll
        for (int ni = 0; ni < 2; ni++) {
          acc[mi][ni] = __builtin_amdgcn_wmma_f32_16x16x32_bf16(
              false, af[mi].v, false, bf2[ni].v, (short)0, acc[mi][ni], false,
              false);
        }
      }
    }
    __syncthreads();   // all reads of buf[cur] done before it is refilled
  }
#undef ISSUE_TILE

  // Epilogue: C/D layout: VGPR j -> row (half*8 + j), col lr within each tile.
#pragma unroll
  for (int mi = 0; mi < 4; mi++) {
#pragma unroll
    for (int ni = 0; ni < 2; ni++) {
      const int row0 = bm + wm + mi * 16 + half * 8;
      const int col  = bn + wn + ni * 16 + lr;
#pragma unroll
      for (int j = 0; j < 8; j++) {
        C[(long)(row0 + j) * N + col] = acc[mi][ni][j];
      }
    }
  }
}

// ---------------------------------------------------------------------------
// Causal depthwise conv (k=4) + SiLU. xi = first DI columns of xz.
// ---------------------------------------------------------------------------
__global__ void conv_silu_kernel(const float* __restrict__ xz,
                                 const float* __restrict__ cw,
                                 const float* __restrict__ cb,
                                 float* __restrict__ xc) {
  long idx = (long)blockIdx.x * blockDim.x + threadIdx.x;   // (b*L+t)*DI + d
  if (idx >= (long)BL * DI) return;
  int d   = (int)(idx % DI);
  long bt = idx / DI;
  int t   = (int)(bt % SEQ);
  long b0 = bt - t;                                          // b*L
  float acc = cb[d];
#pragma unroll
  for (int j = 0; j < 4; j++) {
    int ts = t - 3 + j;
    if (ts >= 0) acc = fmaf(cw[d * 4 + j], xz[(b0 + ts) * (2 * DI) + d], acc);
  }
  xc[idx] = acc / (1.f + __expf(-acc));                      // SiLU
}

// ---------------------------------------------------------------------------
// xp = xconv @ W_x  (N=33: too narrow for WMMA; VALU dot products, ~1 GFLOP)
// Threads in a group of 33 share one row (broadcast) with coalesced W_x reads.
// ---------------------------------------------------------------------------
__global__ void xp_kernel(const float* __restrict__ xc,
                          const float* __restrict__ Wx,
                          float* __restrict__ xp) {
  long idx = (long)blockIdx.x * blockDim.x + threadIdx.x;
  if (idx >= (long)BL * 33) return;
  int  j = (int)(idx % 33);
  long r = idx / 33;
  const float* xr = xc + r * DI;
  float acc = 0.f;
#pragma unroll 4
  for (int k = 0; k < DI; k++) acc = fmaf(xr[k], Wx[k * 33 + j], acc);
  xp[idx] = acc;
}

// ---------------------------------------------------------------------------
// Selective scan, parallel over (b, d, s): 16 lanes own the 16 independent
// state recurrences of one (b,d); y_t reduced with 4 xor-shuffles. Fuses the
// D-skip, SiLU(z) gating, and bf16 output for the final GEMM.
// ---------------------------------------------------------------------------
__global__ __launch_bounds__(256) void scan_kernel(
    const float* __restrict__ xc, const float* __restrict__ xp,
    const float* __restrict__ xz,
    const float* __restrict__ Wdt, const float* __restrict__ bdt,
    const float* __restrict__ Alog, const float* __restrict__ Dsk,
    unsigned short* __restrict__ yb) {
  int tid = blockIdx.x * 256 + threadIdx.x;
  int s  = tid & 15;
  int dg = tid >> 4;             // b*DI + d
  int d  = dg & (DI - 1);
  int b  = dg >> 11;
  float Aa   = -__expf(Alog[s]);
  float wdt  = Wdt[d];
  float bdt_ = bdt[d];
  float Dd   = Dsk[d];
  const float* xpb = xp + (long)b * SEQ * 33;
  float h = 0.f;
  for (int t = 0; t < SEQ; t++) {
    long rbt = (long)b * SEQ + t;
    const float* xpt = xpb + t * 33;
    float v  = fmaf(xpt[0], wdt, bdt_);
    float dt = v > 20.f ? v : log1pf(__expf(v));             // softplus
    float Bv = xpt[1 + s];
    float Cv = xpt[17 + s];
    float xv = xc[rbt * DI + d];
    float Ab = __expf(dt * Aa);
    h = fmaf(Ab, h, dt * Bv * xv);
    float part = h * Cv;
    part += __shfl_xor(part, 1, 16);
    part += __shfl_xor(part, 2, 16);
    part += __shfl_xor(part, 4, 16);
    part += __shfl_xor(part, 8, 16);
    if (s == 0) {
      float y = part + Dd * xv;
      float z = xz[rbt * (2 * DI) + DI + d];
      y *= z / (1.f + __expf(-z));                           // * SiLU(z)
      yb[rbt * DI + d] = f2bf(y);
    }
  }
}

// ---------------------------------------------------------------------------
// Host-side orchestration
// ---------------------------------------------------------------------------
extern "C" void kernel_launch(void* const* d_in, const int* in_sizes, int n_in,
                              void* d_out, int out_size, void* d_ws, size_t ws_size,
                              hipStream_t stream) {
  (void)in_sizes; (void)n_in; (void)out_size; (void)ws_size;
  const float* x      = (const float*)d_in[0];   // (4,2048,1024)
  const float* W_in   = (const float*)d_in[1];   // (1024,4096)
  const float* conv_w = (const float*)d_in[2];   // (2048,1,4)
  const float* conv_b = (const float*)d_in[3];   // (2048,)
  const float* W_x    = (const float*)d_in[4];   // (2048,33)
  const float* W_dt   = (const float*)d_in[5];   // (2048,1)
  const float* b_dt   = (const float*)d_in[6];   // (2048,)
  const float* A_log  = (const float*)d_in[7];   // (16,)
  const float* Dsk    = (const float*)d_in[8];   // (2048,)
  const float* W_out  = (const float*)d_in[9];   // (2048,1024)
  float* out = (float*)d_out;                    // (4,2048,1024)

  char* ws = (char*)d_ws;
  size_t off = 0;
  auto wsAlloc = [&](size_t bytes) -> void* {
    void* p = ws + off;
    off += (bytes + 255) & ~(size_t)255;
    return p;
  };
  unsigned short* Xb    = (unsigned short*)wsAlloc((size_t)BL * DM * 2);      // 16 MB
  unsigned short* Wint  = (unsigned short*)wsAlloc((size_t)(2 * DI) * DM * 2);// 8 MB (4096x1024)
  unsigned short* Woutt = (unsigned short*)wsAlloc((size_t)DM * DI * 2);      // 4 MB (1024x2048)
  float*          xzbuf = (float*)wsAlloc((size_t)BL * (2 * DI) * 4);         // 134 MB
  float*          xconv = (float*)wsAlloc((size_t)BL * DI * 4);               // 67 MB
  float*          xpbuf = (float*)wsAlloc((size_t)BL * 33 * 4);               // ~1 MB
  unsigned short* yb    = (unsigned short*)wsAlloc((size_t)BL * DI * 2);      // 33.5 MB

  // 1) operand conversion / weight transposes (done every call; cheap)
  {
    long n = (long)BL * DM;
    cvt_bf16_kernel<<<(int)((n + 255) / 256), 256, 0, stream>>>(x, Xb, n);
  }
  {
    long n = (long)DM * (2 * DI);
    transpose_bf16_kernel<<<(int)((n + 255) / 256), 256, 0, stream>>>(W_in, Wint, DM, 2 * DI);
  }
  {
    long n = (long)DI * DM;
    transpose_bf16_kernel<<<(int)((n + 255) / 256), 256, 0, stream>>>(W_out, Woutt, DI, DM);
  }

  // 2) xz = x @ W_in   (8192 x 4096 x K=1024)
  {
    dim3 grid((2 * DI) / BNT, BL / BMT);
    gemm_bf16_wmma_kernel<<<grid, 256, 0, stream>>>(Xb, Wint, xzbuf, BL, 2 * DI, DM);
  }

  // 3) depthwise conv + SiLU
  {
    long n = (long)BL * DI;
    conv_silu_kernel<<<(int)(n / 256), 256, 0, stream>>>(xzbuf, conv_w, conv_b, xconv);
  }

  // 4) xp = xconv @ W_x
  {
    long n = (long)BL * 33;
    xp_kernel<<<(int)((n + 255) / 256), 256, 0, stream>>>(xconv, W_x, xpbuf);
  }

  // 5) selective scan + gating -> y (bf16)
  {
    int n = BL * NSTATE;   // 131072 threads
    scan_kernel<<<n / 256, 256, 0, stream>>>(xconv, xpbuf, xzbuf, W_dt, b_dt,
                                             A_log, Dsk, yb);
  }

  // 6) out = y @ W_out   (8192 x 1024 x K=2048)
  {
    dim3 grid(DM / BNT, BL / BMT);
    gemm_bf16_wmma_kernel<<<grid, 256, 0, stream>>>(yb, Woutt, out, BL, DM, DI);
  }
}
